// BertSelfAttention_23089744183834
// MI455X (gfx1250) — matrix operand order repaired and checked by reading, not compiled, not run on hardware
//
#include <hip/hip_runtime.h>
#include <hip/hip_bf16.h>

// ---------------------------------------------------------------------------
// BERT self-attention for MI455X (gfx1250), wave32 + WMMA bf16.
//   B=4, S=2048, H=1024, heads=16, d=64.
// Stage 1: f32 -> bf16 conversion of X and Wq/Wk/Wv.
// Stage 2: QKV GEMM (y = x W^T + b); 32x64 tile per wave, software-pipelined,
//          8 WMMAs per K=32 step sharing 4 B fragments.
// Stage 3: flash-attention; K/V blocks staged into LDS cooperatively
//          (double-buffered, ASYNCcnt-tracked async-to-LDS copies),
//          online softmax, WMMA for QK^T and PV.
// Workspace: Xb 16MB | Wb 6MB | Qb 16MB | Kb 16MB | Vtb 16MB  (~70 MB)
// ---------------------------------------------------------------------------

#define BATCH 4
#define SEQ   2048
#define HID   1024
#define HEADS 16
#define HDIM  64
#define MROWS (BATCH * SEQ)   // 8192

typedef __attribute__((ext_vector_type(16))) __bf16 bf16x16;
typedef __attribute__((ext_vector_type(8)))  float  f32x8;

union BFrag {
  bf16x16 v;
  uint4   q[2];
};

__device__ __forceinline__ unsigned short f2bf(float x) {
  unsigned u = __float_as_uint(x);
  unsigned r = u + 0x7fffu + ((u >> 16) & 1u);   // round-to-nearest-even
  return (unsigned short)(r >> 16);
}

#define WMMA_BF16(A, B, C) \
  __builtin_amdgcn_wmma_f32_16x16x32_bf16(false, (A), false, (B), (short)0, (C), false, false)

// CDNA5 async global->LDS (ASYNCcnt-tracked) — probe-confirmed on this toolchain.
#if __has_builtin(__builtin_amdgcn_global_load_async_to_lds_b128) && \
    __has_builtin(__builtin_amdgcn_s_wait_asynccnt)
#define ASYNC_LDS 1
typedef int v4i_ __attribute__((vector_size(16)));
typedef __attribute__((address_space(1))) v4i_ gv4i;   // global (AS1) 16B vector
typedef __attribute__((address_space(3))) v4i_ lv4i;   // LDS    (AS3) 16B vector
#else
#define ASYNC_LDS 0
#endif

// ---------------------------------------------------------------------------
// Stage 1: f32 -> bf16, 4 elements per thread.
// ---------------------------------------------------------------------------
__global__ void __launch_bounds__(256) cvt_f32_bf16x4(
    const float4* __restrict__ src, unsigned long long* __restrict__ dst, int n4) {
  int i = blockIdx.x * blockDim.x + threadIdx.x;
  if (i < n4) {
    float4 f = src[i];
    unsigned long long r =
        (unsigned long long)f2bf(f.x)         |
        ((unsigned long long)f2bf(f.y) << 16) |
        ((unsigned long long)f2bf(f.z) << 32) |
        ((unsigned long long)f2bf(f.w) << 48);
    dst[i] = r;
  }
}

// ---------------------------------------------------------------------------
// Stage 2: QKV projection. blockIdx.z = 0(Q)/1(K)/2(V).
// One wave -> 32(M) x 64(N) tile: two M row-groups share 4 B fragments,
// 8 WMMAs per K=32 step. K-loop software-pipelined 2 deep.
// ---------------------------------------------------------------------------
__global__ void __launch_bounds__(256) qkv_gemm(
    const unsigned short* __restrict__ Xb,
    const unsigned short* __restrict__ Wb,
    const float* __restrict__ bq,
    const float* __restrict__ bk,
    const float* __restrict__ bv,
    unsigned short* __restrict__ Qb,
    unsigned short* __restrict__ Kb,
    unsigned short* __restrict__ Vtb) {
  const int mode = blockIdx.z;
  const unsigned short* W = Wb + (size_t)mode * HID * HID;
  const float* bias = (mode == 0) ? bq : (mode == 1) ? bk : bv;
  unsigned short* out = (mode == 0) ? Qb : (mode == 1) ? Kb : Vtb;

  const int wave = threadIdx.x >> 5;
  const int lane = threadIdx.x & 31;
  const int half = lane >> 4;
  const int ln   = lane & 15;

  const int tile = blockIdx.x * 8 + wave;   // 4096 wave-tiles (256 Mt x 16 Nt)
  const int m0 = (tile >> 4) * 32;
  const int n0 = (tile & 15) * 64;

  f32x8 accL[4], accH[4];
#pragma unroll
  for (int t = 0; t < 4; ++t)
#pragma unroll
    for (int r = 0; r < 8; ++r) { accL[t][r] = 0.f; accH[t][r] = 0.f; }

  const unsigned short* arowL = Xb + (size_t)(m0 + ln) * HID + half * 8;
  const unsigned short* arowH = arowL + (size_t)16 * HID;
  const unsigned short* brow0 = W  + (size_t)(n0 + ln) * HID + half * 8;

#define LOAD_AF(dst, ROW, K0)                             \
  do {                                                    \
    const unsigned short* ap_ = (ROW) + (K0);             \
    (dst).q[0] = *(const uint4*)(ap_);                    \
    (dst).q[1] = *(const uint4*)(ap_ + 16);               \
  } while (0)
#define LOAD_BF(dst, T, K0)                               \
  do {                                                    \
    const unsigned short* bp_ = brow0 + (size_t)(T) * 16 * HID + (K0); \
    (dst).q[0] = *(const uint4*)(bp_);                    \
    (dst).q[1] = *(const uint4*)(bp_ + 16);               \
  } while (0)

  BFrag aL0, aH0, aL1, aH1, b0[4], b1[4];
  LOAD_AF(aL0, arowL, 0);
  LOAD_AF(aH0, arowH, 0);
#pragma unroll
  for (int t = 0; t < 4; ++t) LOAD_BF(b0[t], t, 0);

  for (int k0 = 0; k0 < HID; k0 += 64) {
    // stage k0+32 while computing k0
    LOAD_AF(aL1, arowL, k0 + 32);
    LOAD_AF(aH1, arowH, k0 + 32);
#pragma unroll
    for (int t = 0; t < 4; ++t) LOAD_BF(b1[t], t, k0 + 32);
#pragma unroll
    for (int t = 0; t < 4; ++t) {
      accL[t] = WMMA_BF16(aL0.v, b0[t].v, accL[t]);
      accH[t] = WMMA_BF16(aH0.v, b0[t].v, accH[t]);
    }
    // stage k0+64 while computing k0+32
    if (k0 + 64 < HID) {
      LOAD_AF(aL0, arowL, k0 + 64);
      LOAD_AF(aH0, arowH, k0 + 64);
#pragma unroll
      for (int t = 0; t < 4; ++t) LOAD_BF(b0[t], t, k0 + 64);
    }
#pragma unroll
    for (int t = 0; t < 4; ++t) {
      accL[t] = WMMA_BF16(aL1.v, b1[t].v, accL[t]);
      accH[t] = WMMA_BF16(aH1.v, b1[t].v, accH[t]);
    }
  }
#undef LOAD_AF
#undef LOAD_BF

  // Epilogue: bias, (Q) softmax scale, bf16 store into attention layouts.
  const float qscale = (mode == 0) ? 0.125f : 1.0f;   // 1/sqrt(64)
#pragma unroll
  for (int t = 0; t < 4; ++t) {
    const int n = n0 + 16 * t + ln;
    const int h = n >> 6;
    const int d = n & 63;
    const float bv_ = bias[n];
#pragma unroll
    for (int g = 0; g < 2; ++g) {
#pragma unroll
      for (int r = 0; r < 8; ++r) {
        const int m = m0 + g * 16 + r + half * 8;
        const int b = m >> 11;
        const int s = m & (SEQ - 1);
        const float acv = g ? accH[t][r] : accL[t][r];
        const float val = (acv + bv_) * qscale;
        size_t addr;
        if (mode == 2)  // V transposed: [B,h,64,S]
          addr = ((size_t)(b * HEADS + h) * HDIM + d) * SEQ + s;
        else            // Q/K: [B,h,S,64]
          addr = ((size_t)(b * HEADS + h) * SEQ + s) * HDIM + d;
        out[addr] = f2bf(val);
      }
    }
  }
}

// ---------------------------------------------------------------------------
// Stage 3: flash attention. 8 waves/block share one (b,h); each wave owns 16
// query rows. K/V 64-key blocks staged into LDS cooperatively, double-buffered.
// grid = (SEQ/128, B*HEADS), block = 256.
// ---------------------------------------------------------------------------
__global__ void __launch_bounds__(256) flash_attn(
    const unsigned short* __restrict__ Qb,
    const unsigned short* __restrict__ Kb,
    const unsigned short* __restrict__ Vtb,
    const float* __restrict__ mask,          // [B,1,1,S]
    float* __restrict__ out) {               // [B,S,H]
  __shared__ __align__(16) unsigned short Klds[2 * 64 * HDIM];  // 16 KB
  __shared__ __align__(16) unsigned short Vlds[2 * 64 * HDIM];  // 16 KB
  __shared__ __align__(16) unsigned short Plds[8 * 16 * HDIM];  // 16 KB

  const int tid  = threadIdx.x;
  const int wave = tid >> 5;
  const int lane = tid & 31;
  const int half = lane >> 4;
  const int ln   = lane & 15;

  const int q0 = blockIdx.x * 128 + wave * 16;
  const int bh = blockIdx.y;
  const int b  = bh >> 4;
  const int h  = bh & 15;

  const unsigned short* Qbase = Qb  + (size_t)bh * SEQ * HDIM;
  const unsigned short* Kbase = Kb  + (size_t)bh * SEQ * HDIM;
  const unsigned short* Vbase = Vtb + (size_t)bh * HDIM * SEQ;
  const float* mrow_g = mask + (size_t)b * SEQ;

  // Cooperative stage of one 64-key K block + V block into LDS buffer `buf`.
  auto stage_kv = [&](int buf, int j0) {
#pragma unroll
    for (int i = 0; i < 2; ++i) {
      const int e   = tid + i * 256;   // 0..511
      const int row = e >> 3;
      const int c8  = (e & 7) * 8;
      const unsigned short* kg = Kbase + (size_t)(j0 + row) * HDIM + c8;
      const unsigned short* vg = Vbase + (size_t)row * SEQ + j0 + c8;
      unsigned short* kl = Klds + buf * 64 * HDIM + row * HDIM + c8;
      unsigned short* vl = Vlds + buf * 64 * HDIM + row * HDIM + c8;
#if ASYNC_LDS
      __builtin_amdgcn_global_load_async_to_lds_b128(
          (gv4i*)kg, (lv4i*)kl, 0, 0);
      __builtin_amdgcn_global_load_async_to_lds_b128(
          (gv4i*)vg, (lv4i*)vl, 0, 0);
#else
      *(uint4*)kl = *(const uint4*)kg;
      *(uint4*)vl = *(const uint4*)vg;
#endif
    }
  };

  // Q fragments for the two K=32 steps of d=64 (A layout, row = ln).
  BFrag qf[2];
  {
    const unsigned short* qr = Qbase + (size_t)(q0 + ln) * HDIM;
#pragma unroll
    for (int ks = 0; ks < 2; ++ks) {
      const unsigned short* p = qr + ks * 32 + half * 8;
      qf[ks].q[0] = *(const uint4*)(p);
      qf[ks].q[1] = *(const uint4*)(p + 16);
    }
  }

  f32x8 ctx[4];
#pragma unroll
  for (int t = 0; t < 4; ++t)
#pragma unroll
    for (int r = 0; r < 8; ++r) ctx[t][r] = 0.f;

  float mrow[8], lrow[8];
#pragma unroll
  for (int r = 0; r < 8; ++r) { mrow[r] = -3.0e38f; lrow[r] = 0.f; }

  unsigned short* pw = Plds + wave * 16 * HDIM;

  stage_kv(0, 0);   // prologue

  for (int j0 = 0; j0 < SEQ; j0 += 64) {
    const int buf = (j0 >> 6) & 1;

    // staging of `buf` complete; all waves done reading buf^1
#if ASYNC_LDS
    __builtin_amdgcn_s_wait_asynccnt(0);
#endif
    __syncthreads();

    if (j0 + 64 < SEQ) stage_kv(buf ^ 1, j0 + 64);
    if (j0 + 128 < SEQ)
      __builtin_prefetch(Kbase + (size_t)(j0 + 128 + (tid & 63)) * HDIM, 0, 1);

    const unsigned short* kl = Klds + buf * 64 * HDIM;
    const unsigned short* vl = Vlds + buf * 64 * HDIM;

    // ---- scores: S = Q K^T (Q pre-scaled by 1/sqrt(d)) ----
    f32x8 sc[4];
#pragma unroll
    for (int t = 0; t < 4; ++t) {
#pragma unroll
      for (int r = 0; r < 8; ++r) sc[t][r] = 0.f;
#pragma unroll
      for (int ks = 0; ks < 2; ++ks) {
        BFrag kf;
        const unsigned short* kp = kl + (16 * t + ln) * HDIM + ks * 32 + half * 8;
        kf.q[0] = *(const uint4*)(kp);
        kf.q[1] = *(const uint4*)(kp + 16);
        sc[t] = WMMA_BF16(qf[ks].v, kf.v, sc[t]);
      }
      const float mv = mrow_g[j0 + 16 * t + ln];   // additive mask per key
#pragma unroll
      for (int r = 0; r < 8; ++r) sc[t][r] += mv;
    }

    // ---- online softmax ----
    float nm[8];
#pragma unroll
    for (int r = 0; r < 8; ++r) {
      float v = fmaxf(fmaxf(sc[0][r], sc[1][r]), fmaxf(sc[2][r], sc[3][r]));
      v = fmaxf(v, __shfl_xor(v, 8, 32));
      v = fmaxf(v, __shfl_xor(v, 4, 32));
      v = fmaxf(v, __shfl_xor(v, 2, 32));
      v = fmaxf(v, __shfl_xor(v, 1, 32));
      nm[r] = v;
    }

    float alpha[8];
#pragma unroll
    for (int r = 0; r < 8; ++r) {
      const float newm = fmaxf(mrow[r], nm[r]);
      alpha[r] = __expf(mrow[r] - newm);
      mrow[r] = newm;
    }

#pragma unroll
    for (int t = 0; t < 4; ++t)
#pragma unroll
      for (int r = 0; r < 8; ++r) sc[t][r] = __expf(sc[t][r] - mrow[r]);

#pragma unroll
    for (int r = 0; r < 8; ++r) {
      float s = sc[0][r] + sc[1][r] + sc[2][r] + sc[3][r];
      s += __shfl_xor(s, 8, 32);
      s += __shfl_xor(s, 4, 32);
      s += __shfl_xor(s, 2, 32);
      s += __shfl_xor(s, 1, 32);
      lrow[r] = lrow[r] * alpha[r] + s;
    }

#pragma unroll
    for (int t = 0; t < 4; ++t)
#pragma unroll
      for (int r = 0; r < 8; ++r) ctx[t][r] *= alpha[r];

    // ---- P (C-layout) -> LDS -> A-layout bf16 fragments (same wave) ----
#pragma unroll
    for (int t = 0; t < 4; ++t)
#pragma unroll
      for (int r = 0; r < 8; ++r)
        pw[(r + half * 8) * HDIM + 16 * t + ln] = f2bf(sc[t][r]);

    // ---- ctx += P V ----
#pragma unroll
    for (int ks = 0; ks < 2; ++ks) {
      BFrag pf;
      const unsigned short* pp = pw + ln * HDIM + ks * 32 + half * 8;
      pf.q[0] = *(const uint4*)(pp);
      pf.q[1] = *(const uint4*)(pp + 16);
#pragma unroll
      for (int tc = 0; tc < 4; ++tc) {
        BFrag vf;
        const unsigned short* vp = vl + (16 * tc + ln) * HDIM + ks * 32 + half * 8;
        vf.q[0] = *(const uint4*)(vp);
        vf.q[1] = *(const uint4*)(vp + 16);
        ctx[tc] = WMMA_BF16(pf.v, vf.v, ctx[tc]);
      }
    }
  }

  // ---- normalize and store f32 output [B,S,H] ----
#pragma unroll
  for (int r = 0; r < 8; ++r) {
    const float inv = 1.0f / lrow[r];
    const int m = q0 + r + half * 8;
    float* orow = out + ((size_t)(b * SEQ + m)) * HID + h * HDIM;
#pragma unroll
    for (int t = 0; t < 4; ++t)
      orow[16 * t + ln] = ctx[t][r] * inv;
  }
}

// ---------------------------------------------------------------------------
extern "C" void kernel_launch(void* const* d_in, const int* in_sizes, int n_in,
                              void* d_out, int out_size, void* d_ws, size_t ws_size,
                              hipStream_t stream) {
  const float* X    = (const float*)d_in[0];
  const float* mask = (const float*)d_in[1];
  const float* Wq   = (const float*)d_in[2];
  const float* bq   = (const float*)d_in[3];
  const float* Wk   = (const float*)d_in[4];
  const float* bk   = (const float*)d_in[5];
  const float* Wv   = (const float*)d_in[6];
  const float* bv   = (const float*)d_in[7];
  float* out = (float*)d_out;

  char* ws = (char*)d_ws;
  const size_t szXb = (size_t)MROWS * HID * 2;
  const size_t szW1 = (size_t)HID * HID * 2;
  unsigned short* Xb  = (unsigned short*)(ws);
  unsigned short* Wb  = (unsigned short*)(ws + szXb);
  unsigned short* Qb  = (unsigned short*)(ws + szXb + 3 * szW1);
  unsigned short* Kb  = (unsigned short*)(ws + szXb + 3 * szW1 + szXb);
  unsigned short* Vtb = (unsigned short*)(ws + szXb + 3 * szW1 + 2 * szXb);

  const int nX4 = MROWS * HID / 4;
  cvt_f32_bf16x4<<<(nX4 + 255) / 256, 256, 0, stream>>>(
      (const float4*)X, (unsigned long long*)Xb, nX4);
  const int nW4 = HID * HID / 4;
  cvt_f32_bf16x4<<<(nW4 + 255) / 256, 256, 0, stream>>>(
      (const float4*)Wq, (unsigned long long*)(Wb), nW4);
  cvt_f32_bf16x4<<<(nW4 + 255) / 256, 256, 0, stream>>>(
      (const float4*)Wk, (unsigned long long*)(Wb + HID * HID), nW4);
  cvt_f32_bf16x4<<<(nW4 + 255) / 256, 256, 0, stream>>>(
      (const float4*)Wv, (unsigned long long*)(Wb + 2 * HID * HID), nW4);

  // 256 Mtiles(32) x 16 Ntiles(64) = 4096 wave-tiles -> 512 blocks of 8 waves.
  qkv_gemm<<<dim3(512, 1, 3), 256, 0, stream>>>(Xb, Wb, bq, bk, bv, Qb, Kb, Vtb);

  flash_attn<<<dim3(SEQ / 128, BATCH * HEADS), 256, 0, stream>>>(Qb, Kb, Vtb, mask, out);
}
